// RawGATEncoder_39247411151304
// MI455X (gfx1250) — matrix operand order repaired and checked by reading, not compiled, not run on hardware
//
#include <hip/hip_runtime.h>
#include <hip/hip_bf16.h>
#include <stdint.h>

// ---------------------------------------------------------------------------
// 3-layer GAT on MI455X (gfx1250, wave32).
// Dense projections via v_wmma_f32_16x16x32_bf16 (f32 accumulate).
// Sparse attention/aggregation via L2-resident float atomics (xp = 102MB < 192MB L2).
// ---------------------------------------------------------------------------

#define TPB 256
#define HEADS 4

typedef __attribute__((ext_vector_type(16))) __bf16 v16bf;
typedef __attribute__((ext_vector_type(8)))  float  v8f;

union Frag32 { uint4 q[2]; v16bf v; };   // 32 bytes = 8 VGPRs per lane

__device__ __forceinline__ unsigned short f2bf(float f) {
  unsigned u = __float_as_uint(f);
  u += 0x7FFFu + ((u >> 16) & 1u);        // round-to-nearest-even
  return (unsigned short)(u >> 16);
}
// order-preserving float->uint key (for atomicMax-based segment max)
__device__ __forceinline__ unsigned fkey(float f) {
  unsigned u = __float_as_uint(f);
  return (u & 0x80000000u) ? ~u : (u | 0x80000000u);
}
__device__ __forceinline__ float funkey(unsigned k) {
  unsigned b = (k & 0x80000000u) ? (k & 0x7FFFFFFFu) : ~k;
  return __uint_as_float(b);
}

// ---------------- W packing: row-major [K,N] f32 -> WMMA B-fragment bf16 ----
// B 32x16 bf16 layout: lane (0..15) holds column lane, K={0..7,16..23};
// lanes 16..31 hold column lane-16, K={8..15,24..31}. One thread emits one lane-chunk
// (16 contiguous bf16 = 32B) so the GEMM can load it with two b128 loads.
__global__ void pack_w_kernel(const float* __restrict__ W,
                              unsigned short* __restrict__ out,
                              int K, int Nn) {
  int KT = K >> 5, NT = Nn >> 4;
  int t = blockIdx.x * blockDim.x + threadIdx.x;
  if (t >= NT * KT * 32) return;
  int lane = t & 31;
  int kt   = (t >> 5) % KT;
  int nt   = (t >> 5) / KT;
  int col  = lane & 15;
  int kb   = (lane >> 4) << 3;
  unsigned short* o = out + (size_t)t * 16;
#pragma unroll
  for (int j = 0; j < 16; ++j) {
    int k = kt * 32 + kb + (j & 7) + ((j >> 3) << 4);
    o[j] = f2bf(W[(size_t)k * Nn + nt * 16 + col]);
  }
}

__global__ void cvt_bf16_kernel(const float* __restrict__ in,
                                unsigned short* __restrict__ out, size_t n) {
  size_t i = (size_t)blockIdx.x * blockDim.x + threadIdx.x;
  if (i < n) out[i] = f2bf(in[i]);
}

// ---------------- WMMA GEMM: XP[M,Nout] = Abf16[M,K] @ Bpk ------------------
// One wave per 16-row tile; all A fragments held in VGPRs across the N sweep.
template <int KT>
__global__ __launch_bounds__(TPB) void gemm_wmma_kernel(
    const unsigned short* __restrict__ A,    // [M, K] bf16 row-major
    const unsigned short* __restrict__ Bpk,  // packed fragments
    float* __restrict__ C,                   // [M, ldc] f32
    int MT, int NT, int ldc) {
  const int K = KT * 32;
  int wave = blockIdx.x * (blockDim.x >> 5) + (threadIdx.x >> 5);
  if (wave >= MT) return;
  int lane = threadIdx.x & 31;
  int r    = lane & 15;        // A row within tile / C column within tile
  int hi   = lane >> 4;

  // A 16x32 bf16 layout: lanes 0..15 row=lane K={0..7,16..23};
  //                      lanes 16..31 same rows, K={8..15,24..31}.
  Frag32 a[KT];
  size_t aBase = (size_t)(wave * 16 + r) * K + hi * 8;
#pragma unroll
  for (int kt = 0; kt < KT; ++kt) {
    const uint4* p = reinterpret_cast<const uint4*>(A + aBase + kt * 32);
    a[kt].q[0] = p[0];          // K k0..k0+7
    a[kt].q[1] = p[2];          // K k0+16..k0+23 (+32 bytes)
  }

  for (int nt = 0; nt < NT; ++nt) {
    v8f acc = {};
#pragma unroll
    for (int kt = 0; kt < KT; ++kt) {
      const uint4* bp = reinterpret_cast<const uint4*>(
          Bpk + ((size_t)(nt * KT + kt) * 32 + lane) * 16);
      Frag32 b;
      b.q[0] = bp[0];
      b.q[1] = bp[1];
      acc = __builtin_amdgcn_wmma_f32_16x16x32_bf16(
          false, a[kt].v, false, b.v, (short)0, acc, false, false);
    }
    // C/D layout: VGPR v -> row v + 8*hi, column = lane&15
    size_t cb = (size_t)(wave * 16 + hi * 8) * ldc + nt * 16 + r;
#pragma unroll
    for (int v = 0; v < 8; ++v) C[cb + (size_t)v * ldc] = acc[v];
  }
}

// ---------------- attention scalars: al[n,h] = <xp[n,h,:], a[h,:]> ----------
__global__ void attn_al_kernel(const float* __restrict__ XP,
                               const float* __restrict__ as,
                               const float* __restrict__ ad,
                               float* __restrict__ ALs, float* __restrict__ ALd,
                               int N, int Cc) {
  int t = blockIdx.x * blockDim.x + threadIdx.x;
  if (t >= N * HEADS) return;
  int n = t / HEADS, h = t % HEADS;
  const float* xp = XP + ((size_t)n * HEADS + h) * Cc;
  float s = 0.f, d = 0.f;
  for (int c = 0; c < Cc; ++c) {
    float v = xp[c];
    s += v * as[h * Cc + c];
    d += v * ad[h * Cc + c];
  }
  ALs[t] = s;
  ALd[t] = d;
}

__global__ void init_bias_kernel(float* __restrict__ O, const float* __restrict__ b,
                                 size_t n, int F) {
  size_t i = (size_t)blockIdx.x * blockDim.x + threadIdx.x;
  if (i < n) O[i] = b[i % F];
}
__global__ void fill_u32_kernel(unsigned* __restrict__ p, unsigned v, size_t n) {
  size_t i = (size_t)blockIdx.x * blockDim.x + threadIdx.x;
  if (i < n) p[i] = v;
}

// ---------------- edge phase ------------------------------------------------
__global__ void edge_logits_kernel(const int* __restrict__ src, const int* __restrict__ dst,
                                   const float* __restrict__ ALs, const float* __restrict__ ALd,
                                   float* __restrict__ EB, unsigned* __restrict__ MX,
                                   int E, int N) {
  int e = blockIdx.x * blockDim.x + threadIdx.x;
  if (e >= E + N) return;
  int s = (e < E) ? src[e] : (e - E);
  int d = (e < E) ? dst[e] : (e - E);
#pragma unroll
  for (int h = 0; h < HEADS; ++h) {
    float v = ALs[s * HEADS + h] + ALd[d * HEADS + h];
    v = v > 0.f ? v : 0.2f * v;                      // leaky_relu
    EB[(size_t)e * HEADS + h] = v;
    atomicMax(&MX[d * HEADS + h], fkey(v));
  }
}

__global__ void edge_exp_kernel(const int* __restrict__ src, const int* __restrict__ dst,
                                const unsigned* __restrict__ MX,
                                float* __restrict__ EB, float* __restrict__ DEN,
                                int E, int N) {
  int e = blockIdx.x * blockDim.x + threadIdx.x;
  if (e >= E + N) return;
  int d = (e < E) ? dst[e] : (e - E);
#pragma unroll
  for (int h = 0; h < HEADS; ++h) {
    float m = funkey(MX[d * HEADS + h]);
    float p = __expf(EB[(size_t)e * HEADS + h] - m);
    EB[(size_t)e * HEADS + h] = p;
    atomicAdd(&DEN[d * HEADS + h], p);
  }
}

// one thread = one (edge, 8-channel group): float4 gathers + 8 float atomics
__global__ void edge_scatter_kernel(const int* __restrict__ src, const int* __restrict__ dst,
                                    const float* __restrict__ EB, const float* __restrict__ DEN,
                                    const float* __restrict__ XP, float* __restrict__ O,
                                    int E, int N, int F) {
  int Cc = F / HEADS;
  int G  = F >> 3;
  long long t = (long long)blockIdx.x * blockDim.x + threadIdx.x;
  long long total = (long long)(E + N) * G;
  if (t >= total) return;
  int g = (int)(t % G);
  long long e = t / G;
  int s = (e < E) ? src[e] : (int)(e - E);
  int d = (e < E) ? dst[e] : (int)(e - E);
  int h = (g * 8) / Cc;
  float alpha = EB[(size_t)e * HEADS + h] / DEN[(size_t)d * HEADS + h];
  const float4* xp = reinterpret_cast<const float4*>(XP + (size_t)s * F + g * 8);
  float4 x0 = xp[0], x1 = xp[1];
  float* o = O + (size_t)d * F + g * 8;
  atomicAdd(o + 0, alpha * x0.x);
  atomicAdd(o + 1, alpha * x0.y);
  atomicAdd(o + 2, alpha * x0.z);
  atomicAdd(o + 3, alpha * x0.w);
  atomicAdd(o + 4, alpha * x1.x);
  atomicAdd(o + 5, alpha * x1.y);
  atomicAdd(o + 6, alpha * x1.z);
  atomicAdd(o + 7, alpha * x1.w);
}

__global__ void relu_kernel(float* __restrict__ p, size_t n) {
  size_t i = (size_t)blockIdx.x * blockDim.x + threadIdx.x;
  if (i < n) p[i] = fmaxf(p[i], 0.f);
}

// ---------------------------------------------------------------------------
static inline unsigned cdiv(size_t a, size_t b) { return (unsigned)((a + b - 1) / b); }

extern "C" void kernel_launch(void* const* d_in, const int* in_sizes, int n_in,
                              void* d_out, int out_size, void* d_ws, size_t ws_size,
                              hipStream_t stream) {
  const float* x   = (const float*)d_in[0];
  const int*   ei  = (const int*)d_in[1];
  const float* W0  = (const float*)d_in[2];
  const float* as0 = (const float*)d_in[3];
  const float* ad0 = (const float*)d_in[4];
  const float* b0  = (const float*)d_in[5];
  const float* W1  = (const float*)d_in[6];
  const float* as1 = (const float*)d_in[7];
  const float* ad1 = (const float*)d_in[8];
  const float* b1  = (const float*)d_in[9];
  const float* W2  = (const float*)d_in[10];
  const float* as2 = (const float*)d_in[11];
  const float* ad2 = (const float*)d_in[12];
  const float* b2  = (const float*)d_in[13];

  const int N = in_sizes[0] / 128;   // 100000
  const int E = in_sizes[1] / 2;     // 1600000
  const int* src = ei;
  const int* dst = ei + E;

  // workspace carve-out
  char*  ws  = (char*)d_ws;
  size_t off = 0;
  auto carve = [&](size_t bytes) -> void* {
    void* p = ws + off;
    off += (bytes + 255) & ~(size_t)255;
    return p;
  };
  unsigned short* Wpk0 = (unsigned short*)carve((size_t)128 * 256 * 2);
  unsigned short* Wpk1 = (unsigned short*)carve((size_t)256 * 256 * 2);
  unsigned short* Wpk2 = (unsigned short*)carve((size_t)256 * 128 * 2);
  unsigned short* Xb   = (unsigned short*)carve((size_t)N * 256 * 2);
  float*    XP  = (float*)carve((size_t)N * 256 * 4);
  float*    Hb  = (float*)carve((size_t)N * 256 * 4);
  float*    ALs = (float*)carve((size_t)N * HEADS * 4);
  float*    ALd = (float*)carve((size_t)N * HEADS * 4);
  unsigned* MX  = (unsigned*)carve((size_t)N * HEADS * 4);
  float*    DEN = (float*)carve((size_t)N * HEADS * 4);
  float*    EB  = (float*)carve((size_t)(E + N) * HEADS * 4);
  (void)ws_size; (void)n_in; (void)out_size;

  const int MT = N / 16;                 // 6250 row tiles (N divisible by 16)
  const unsigned gemmBlocks = cdiv(MT, 8);
  const unsigned edgeBlocks = cdiv((size_t)(E + N), TPB);
  const unsigned nhBlocks   = cdiv((size_t)N * HEADS, TPB);

  // pack all weights once per call
  pack_w_kernel<<<cdiv((size_t)(256 / 16) * (128 / 32) * 32, TPB), TPB, 0, stream>>>(W0, Wpk0, 128, 256);
  pack_w_kernel<<<cdiv((size_t)(256 / 16) * (256 / 32) * 32, TPB), TPB, 0, stream>>>(W1, Wpk1, 256, 256);
  pack_w_kernel<<<cdiv((size_t)(128 / 16) * (256 / 32) * 32, TPB), TPB, 0, stream>>>(W2, Wpk2, 256, 128);

  // ---- layer: in -> (gemm, attn, softmax, scatter, +bias [,relu]) ----
  auto run_layer = [&](const float* inF, int Fin, int Fout, int Cc,
                       unsigned short* Wpk, const float* as, const float* ad,
                       const float* bias, float* out, bool relu) {
    cvt_bf16_kernel<<<cdiv((size_t)N * Fin, TPB), TPB, 0, stream>>>(inF, Xb, (size_t)N * Fin);
    if (Fin == 128)
      gemm_wmma_kernel<4><<<gemmBlocks, TPB, 0, stream>>>(Xb, Wpk, XP, MT, Fout / 16, Fout);
    else
      gemm_wmma_kernel<8><<<gemmBlocks, TPB, 0, stream>>>(Xb, Wpk, XP, MT, Fout / 16, Fout);
    attn_al_kernel<<<nhBlocks, TPB, 0, stream>>>(XP, as, ad, ALs, ALd, N, Cc);
    init_bias_kernel<<<cdiv((size_t)N * Fout, TPB), TPB, 0, stream>>>(out, bias, (size_t)N * Fout, Fout);
    fill_u32_kernel<<<nhBlocks, TPB, 0, stream>>>(MX, 0u, (size_t)N * HEADS);
    fill_u32_kernel<<<nhBlocks, TPB, 0, stream>>>((unsigned*)DEN, 0u, (size_t)N * HEADS);
    edge_logits_kernel<<<edgeBlocks, TPB, 0, stream>>>(src, dst, ALs, ALd, EB, MX, E, N);
    edge_exp_kernel<<<edgeBlocks, TPB, 0, stream>>>(src, dst, MX, EB, DEN, E, N);
    edge_scatter_kernel<<<cdiv((size_t)(E + N) * (Fout / 8), TPB), TPB, 0, stream>>>(
        src, dst, EB, DEN, XP, out, E, N, Fout);
    if (relu)
      relu_kernel<<<cdiv((size_t)N * Fout, TPB), TPB, 0, stream>>>(out, (size_t)N * Fout);
  };

  run_layer(x,  128, 256, 64, Wpk0, as0, ad0, b0, Hb,           true);
  run_layer(Hb, 256, 256, 64, Wpk1, as1, ad1, b1, Hb,           true);
  run_layer(Hb, 256, 128, 32, Wpk2, as2, ad2, b2, (float*)d_out, false);
}
